// EnhancedLIFWithMemory_61211873902714
// MI455X (gfx1250) — matrix operand order repaired and checked by reading, not compiled, not run on hardware
//
#include <hip/hip_runtime.h>
#include <hip/hip_bf16.h>
#include <cstdint>

// ---------------------------------------------------------------------------
// EnhancedLIFWithMemory on gfx1250:
//   Phase 1: syn[b,t,h] = spikes[b,t,:] @ W[:,h] + b[h]   (bf16 WMMA, f32 acc)
//            -> written directly into d_out (mask is applied post-GEMM, so the
//               T-loop of GEMMs hoists into one [32768,1024]x[1024,1024] GEMM)
//   Phase 2: per-(b,h) sequential LIF recurrence over T, overwriting d_out.
//
// Grid order: blockIdx.x = N-tile (8), blockIdx.y = M-tile (256) so the 8
// blocks sharing one 512KB A row-block launch consecutively -> A streamed from
// HBM once and re-read from the 192MB L2; W (4MB) stays L2-resident.
// ---------------------------------------------------------------------------

typedef __attribute__((ext_vector_type(16))) __bf16 v16bf;
typedef __attribute__((ext_vector_type(2)))  __bf16 v2bf;
typedef __attribute__((ext_vector_type(8)))  float  v8f;
typedef __attribute__((ext_vector_type(4)))  float  f32x4;

#define GEMM_M 32768  // B*T
#define GEMM_N 1024   // H
#define GEMM_K 1024   // D
#define TILE_M 128
#define TILE_N 128
#define TILE_K 32
#define LDSTR  40     // bf16 elems per LDS row (16B-aligned frags, conflict-free)
#define NSLAB  (GEMM_K / TILE_K)

union FragBF { uint4 q[2]; v16bf v; };
union AccF   { v8f  v; float f[8]; };

// packed f32x2 -> bf16x2 via native casts (clang emits packed hw convert)
__device__ __forceinline__ unsigned pack2bf(float lo, float hi) {
    union { v2bf v; unsigned u; } r;
    r.v = (v2bf){ (__bf16)lo, (__bf16)hi };
    return r.u;
}

__global__ __launch_bounds__(256)
void lif_gemm_bf16_wmma(const float* __restrict__ A,     // spikes [32768,1024]
                        const float* __restrict__ W,     // [1024,1024]
                        const float* __restrict__ bias,  // [1024]
                        float* __restrict__ out)         // [32768,1024]
{
    __shared__ __align__(16) unsigned short lsA[2][TILE_M * LDSTR]; // 2 x 10240 B
    __shared__ __align__(16) unsigned short lsB[2][TILE_N * LDSTR]; // 2 x 10240 B (transposed W)

    const int n0   = blockIdx.x * TILE_N;   // x = N tile -> consecutive blocks share A rows
    const int m0   = blockIdx.y * TILE_M;
    const int t    = threadIdx.x;
    const int lane = t & 31;
    const int wave = t >> 5;        // 0..7
    const int wm   = wave >> 1;     // 0..3 -> 32-row strip
    const int wn   = wave & 1;      // 0..1 -> 64-col strip

    // WMMA bf16 fragment addressing (CDNA5 ISA 7.12.2)
    const int frow = lane & 15;
    const int fhk  = (lane >> 4) * 8;    // A: K sub-offset per lane half
    const int bks  = (lane >> 4) * 8;    // B: K sub-offset per lane half

    AccF acc[2][4];
#pragma unroll
    for (int i = 0; i < 2; ++i)
#pragma unroll
        for (int j = 0; j < 4; ++j)
            acc[i][j].v = (v8f){0.f,0.f,0.f,0.f,0.f,0.f,0.f,0.f};

    // A staging: thread -> (row 0..127, k-half 0/16), 4x 128b loads per slab
    const int arow = t >> 1;
    const int acol = (t & 1) * 16;
    const float* gA = A + (size_t)(m0 + arow) * GEMM_K + acol;

    // B staging: thread -> (k-pair 0..15, 8 consecutive n), packed bf16x2 stores
    const int kp   = t >> 4;          // 0..15 -> rows k=2kp, 2kp+1
    const int nseg = (t & 15) * 8;    // 0..120
    const float* gB = W + (size_t)(2 * kp) * GEMM_N + n0 + nseg;

    auto stageA = [&](int buf, int k0) {
        const float* p = gA + k0;
#pragma unroll
        for (int i = 0; i < 4; ++i) {
            f32x4 f = *(const f32x4*)(p + 4 * i);   // temporal: reused via L2 by peer N-blocks
            unsigned* dst = (unsigned*)&lsA[buf][arow * LDSTR + acol + 4 * i];
            dst[0] = pack2bf(f.x, f.y);
            dst[1] = pack2bf(f.z, f.w);
        }
    };
    auto stageB = [&](int buf, int k0) {
        const float* p0 = gB + (size_t)k0 * GEMM_N;  // row k=2kp
        const float* p1 = p0 + GEMM_N;               // row k=2kp+1
        f32x4 a0 = *(const f32x4*)(p0);
        f32x4 a1 = *(const f32x4*)(p0 + 4);
        f32x4 b0 = *(const f32x4*)(p1);
        f32x4 b1 = *(const f32x4*)(p1 + 4);
        float wa[8] = {a0.x,a0.y,a0.z,a0.w,a1.x,a1.y,a1.z,a1.w};
        float wb[8] = {b0.x,b0.y,b0.z,b0.w,b1.x,b1.y,b1.z,b1.w};
#pragma unroll
        for (int i = 0; i < 8; ++i)   // lsB[n][k] = {W[2kp,n], W[2kp+1,n]}
            *(unsigned*)&lsB[buf][(nseg + i) * LDSTR + 2 * kp] = pack2bf(wa[i], wb[i]);
    };

    stageA(0, 0);
    stageB(0, 0);
    __syncthreads();

    for (int slab = 0; slab < NSLAB; ++slab) {
        const int cur = slab & 1;
        const int k0  = slab * TILE_K;

        if (slab + 1 < NSLAB) {               // stage next slab into other buffer
            stageA(cur ^ 1, k0 + TILE_K);
            stageB(cur ^ 1, k0 + TILE_K);
            __builtin_prefetch(gA + k0 + 2 * TILE_K, 0, 3);  // WGP-scope prefetch, all levels
        }

        FragBF afr[2], bfr[4];
#pragma unroll
        for (int i = 0; i < 2; ++i) {
            const unsigned short* base = &lsA[cur][(wm * 32 + i * 16 + frow) * LDSTR];
            afr[i].q[0] = *(const uint4*)(base + fhk);
            afr[i].q[1] = *(const uint4*)(base + 16 + fhk);
        }
#pragma unroll
        for (int j = 0; j < 4; ++j) {
            const unsigned short* base = &lsB[cur][(wn * 64 + j * 16 + frow) * LDSTR];
            bfr[j].q[0] = *(const uint4*)(base + bks);
            bfr[j].q[1] = *(const uint4*)(base + 16 + bks);
        }

#pragma unroll
        for (int i = 0; i < 2; ++i)
#pragma unroll
            for (int j = 0; j < 4; ++j)
                acc[i][j].v = __builtin_amdgcn_wmma_f32_16x16x32_bf16(
                    false, afr[i].v, false, bfr[j].v,
                    (short)0, acc[i][j].v, false, false);

        __syncthreads();
    }

    // epilogue: bias + f32 store (C layout: M = r + 8*(lane>=16), N = lane&15)
    const int cN = lane & 15;
    const int cM = (lane >> 4) * 8;
#pragma unroll
    for (int j = 0; j < 4; ++j) {
        const int gcol = n0 + wn * 64 + j * 16 + cN;
        const float bj = bias[gcol];
#pragma unroll
        for (int i = 0; i < 2; ++i) {
            const int rbase = m0 + wm * 32 + i * 16 + cM;
#pragma unroll
            for (int r = 0; r < 8; ++r)
                out[(size_t)(rbase + r) * GEMM_N + gcol] = acc[i][j].f[r] + bj;
        }
    }
}

__global__ __launch_bounds__(256)
void lif_scan(float* __restrict__ buf,  // in: syn_in(+bias), out: spikes [B,T,H]
              const float* __restrict__ lt_mem, const float* __restrict__ lt_syn,
              const float* __restrict__ lt_ref, const float* __restrict__ lt_adapt)
{
    const int H = 1024, T = 512;
    const int tid = blockIdx.x * blockDim.x + threadIdx.x;  // 0..65535
    const int h   = tid & (H - 1);
    const int b0  = tid >> 10;

    const float a_mem = __expf(-1e-3f / __expf(lt_mem[0]));
    const float a_syn = __expf(-1e-3f / __expf(lt_syn[0]));
    const float a_ad  = __expf(-1e-3f / __expf(lt_adapt[0]));
    const float ref_steps = __expf(lt_ref[0]) * 1e3f;   // tau_ref / DT

    float v = 0.f, isyn = 0.f, refrac = 0.f, adapt = 0.f;
    size_t idx = (size_t)b0 * T * H + h;
    for (int t = 0; t < T; ++t, idx += H) {
        const float syn  = buf[idx];
        const float mask = (refrac <= 0.f) ? 1.f : 0.f;
        const float eff  = 1.0f + adapt;                 // THRESHOLD + adapt
        isyn = a_syn * isyn + syn * mask;
        v    = a_mem * v + isyn;
        const float s = (v - eff > 0.f) ? 1.f : 0.f;     // Heaviside fwd
        v -= s * eff * 0.8f;                             // soft reset
        refrac = (s > 0.f) ? ref_steps : fmaxf(0.f, refrac - 1.f);
        adapt  = a_ad * adapt + s * 0.1f;
        buf[idx] = s;
    }
}

extern "C" void kernel_launch(void* const* d_in, const int* in_sizes, int n_in,
                              void* d_out, int out_size, void* d_ws, size_t ws_size,
                              hipStream_t stream) {
    const float* spikes = (const float*)d_in[0];
    const float* W      = (const float*)d_in[1];
    const float* bias   = (const float*)d_in[2];
    const float* lt_mem = (const float*)d_in[3];
    const float* lt_syn = (const float*)d_in[4];
    const float* lt_ref = (const float*)d_in[5];
    const float* lt_ad  = (const float*)d_in[6];
    float* out = (float*)d_out;

    (void)in_sizes; (void)n_in; (void)out_size; (void)d_ws; (void)ws_size;

    // Phase 1: [32768,1024]x[1024,1024] -> d_out (pre-activation currents)
    // x = N tiles (fastest) so consecutive blocks reuse the same A rows via L2.
    dim3 g1(GEMM_N / TILE_N, GEMM_M / TILE_M);  // 8 x 256
    lif_gemm_bf16_wmma<<<g1, 256, 0, stream>>>(spikes, W, bias, out);

    // Phase 2: per-channel LIF recurrence over T, in place on d_out
    lif_scan<<<(64 * 1024) / 256, 256, 0, stream>>>(out, lt_mem, lt_syn, lt_ref, lt_ad);
}
